// EnSSM_77206332113515
// MI455X (gfx1250) — matrix operand order repaired and checked by reading, not compiled, not run on hardware
//
#include <hip/hip_runtime.h>
#include <math.h>

typedef __attribute__((ext_vector_type(16))) _Float16 v16h;
typedef __attribute__((ext_vector_type(8)))  _Float16 v8h;
typedef __attribute__((ext_vector_type(8)))  float    v8f;

#define LN_EPS 1e-6f
#define BN_EPS 1e-5f
#define LSEQ   1024          // sequence length (fixed by the problem)
#define LSEQ_LOG 10

// ---------------------------------------------------------------------------
// WMMA fragment-order LDS staging helpers.
// A fragment (16x32 f16):  lane = m + 16*((k>>3)&1), elem = (k&7) + 8*(k>>4)
// B fragment (32x16 f16):  lane = n + 16*((k>>3)&1), elem = (k&7) + 8*(k>>4)
// 8 consecutive k (k%8==0) land contiguously -> one b128 LDS store per group.
// ---------------------------------------------------------------------------
__device__ __forceinline__ void store_afrag_group(v16h (*Afrag)[32], int r, int cg,
                                                  const v8h& hv)
{
    _Float16* dst = (_Float16*)&Afrag[r >> 4][(r & 15) + 16 * ((cg >> 3) & 1)];
    *(v8h*)(dst + 8 * (cg >> 4)) = hv;
}
__device__ __forceinline__ void store_bfrag_group(v16h (*Bfrag)[32], int nn, int kg,
                                                  const v8h& hv)
{
    _Float16* dst = (_Float16*)&Bfrag[nn >> 4][(nn & 15) + 16 * ((kg >> 3) & 1)];
    *(v8h*)(dst + 8 * (kg >> 4)) = hv;
}
__device__ __forceinline__ v8h cvt8(const float4& f0, const float4& f1)
{
    v8h h;
    h[0] = (_Float16)f0.x; h[1] = (_Float16)f0.y; h[2] = (_Float16)f0.z; h[3] = (_Float16)f0.w;
    h[4] = (_Float16)f1.x; h[5] = (_Float16)f1.y; h[6] = (_Float16)f1.z; h[7] = (_Float16)f1.w;
    return h;
}
__device__ __forceinline__ v8h zero8()
{
    v8h h;
    #pragma unroll
    for (int j = 0; j < 8; ++j) h[j] = (_Float16)0.0f;
    return h;
}

// ---------------------------------------------------------------------------
// Generic WMMA GEMM:  C[M,N] = epilogue( A[M,K] * W[N,K]^T )
// Block tile 64x64, K-step 32, 128 threads = 4 waves; wave w owns M rows
// [16w,16w+16) x 64 N.  Double-buffered LDS software pipeline: prefetch tile
// k+1 from global into registers while WMMAs consume tile k; one barrier per
// K-step.  AL=true strips all K/N guards (K%32==0, N%64==0).
// EPI: 0 none | 1 BN+ReLU | 2 C += acc | 3 softplus(acc + e0[n])
// ---------------------------------------------------------------------------
template<int EPI, bool AL>
__global__ __launch_bounds__(128)
void gemm_wmma(const float* __restrict__ A, int lda,
               const float* __restrict__ W, int ldw,
               float* __restrict__ C, int ldc,
               int N, int K,
               const float* __restrict__ e0, const float* __restrict__ e1,
               float escale)
{
    __shared__ v16h Afrag[2][4][32];
    __shared__ v16h Bfrag[2][4][32];

    const int tid  = threadIdx.x;
    const int lane = tid & 31;
    const int wave = tid >> 5;

    v8f acc[4];
    #pragma unroll
    for (int nt = 0; nt < 4; ++nt)
        #pragma unroll
        for (int i = 0; i < 8; ++i) acc[nt][i] = 0.0f;

    const int mBlk = blockIdx.x * 64;
    const int nBlk = blockIdx.y * 64;

    v8h ra[2], rb[2];

    auto loadTile = [&](int k0) {
        #pragma unroll
        for (int gg = 0; gg < 2; ++gg) {
            int g  = tid + gg * 128;
            int r  = g >> 2;
            int cg = (g & 3) * 8;
            int kk = k0 + cg;
            if (AL || kk + 8 <= K) {
                const float* p = &A[(mBlk + r) * lda + kk];
                ra[gg] = cvt8(*(const float4*)p, *(const float4*)(p + 4));
            } else {
                #pragma unroll
                for (int j = 0; j < 8; ++j) {
                    int k2 = kk + j;
                    ra[gg][j] = (_Float16)((k2 < K) ? A[(mBlk + r) * lda + k2] : 0.0f);
                }
            }
        }
        #pragma unroll
        for (int gg = 0; gg < 2; ++gg) {
            int g  = tid + gg * 128;
            int nn = g >> 2;
            int kg = (g & 3) * 8;
            int kk = k0 + kg;
            int ng = nBlk + nn;
            if (AL || (ng < N && kk + 8 <= K)) {
                const float* p = &W[ng * ldw + kk];
                rb[gg] = cvt8(*(const float4*)p, *(const float4*)(p + 4));
            } else {
                #pragma unroll
                for (int j = 0; j < 8; ++j) {
                    int k2 = kk + j;
                    rb[gg][j] = (_Float16)((ng < N && k2 < K) ? W[ng * ldw + k2] : 0.0f);
                }
            }
        }
    };
    auto storeTile = [&](int buf) {
        #pragma unroll
        for (int gg = 0; gg < 2; ++gg) {
            int g = tid + gg * 128;
            store_afrag_group(Afrag[buf], g >> 2, (g & 3) * 8, ra[gg]);
        }
        #pragma unroll
        for (int gg = 0; gg < 2; ++gg) {
            int g = tid + gg * 128;
            store_bfrag_group(Bfrag[buf], g >> 2, (g & 3) * 8, rb[gg]);
        }
    };

    const int nk = (K + 31) >> 5;
    loadTile(0);
    storeTile(0);
    __syncthreads();

    for (int kt = 0; kt < nk; ++kt) {
        const int cb = kt & 1;
        const bool hasNext = (kt + 1 < nk);
        if (hasNext) loadTile((kt + 1) << 5);   // overlap with WMMA below

        v16h afrag = Afrag[cb][wave][lane];
        #pragma unroll
        for (int nt = 0; nt < 4; ++nt) {
            v16h bfrag = Bfrag[cb][nt][lane];
            acc[nt] = __builtin_amdgcn_wmma_f32_16x16x32_f16(
                false, afrag, false, bfrag, (short)0, acc[nt], false, false);
        }
        if (hasNext) {
            storeTile(cb ^ 1);
            __syncthreads();
        }
    }

    const int mBase = mBlk + wave * 16 + ((lane >> 4) * 8);
    #pragma unroll
    for (int nt = 0; nt < 4; ++nt) {
        const int n = nBlk + nt * 16 + (lane & 15);
        if (AL || n < N) {
            #pragma unroll
            for (int i = 0; i < 8; ++i) {
                int idx = (mBase + i) * ldc + n;
                float v = acc[nt][i];
                if (EPI == 1) { v = v * e0[n] * escale + e1[n]; v = v > 0.f ? v : 0.f; }
                else if (EPI == 2) { v += C[idx]; }
                else if (EPI == 3) { v += e0[n]; v = (v > 20.f) ? v : log1pf(__expf(v)); }
                C[idx] = v;
            }
        }
    }
}

// ---------------------------------------------------------------------------
// Conv(K=3, pad=1) as implicit-im2col WMMA GEMM + BN + ReLU.
// Contraction K-index = k*Cin + c (k-major); Cin is a power of two and a
// multiple of 32, so each 8-wide group has fixed k and contiguous channels.
// Same double-buffered pipeline as gemm_wmma.  K3%32==0, N%64==0.
// ---------------------------------------------------------------------------
__global__ __launch_bounds__(128)
void conv_gemm_wmma(const float* __restrict__ X, int Cin, int lcsh,
                    const float* __restrict__ Wc,
                    const float* __restrict__ bn_g, const float* __restrict__ bn_b,
                    float* __restrict__ C, int N, float escale)
{
    __shared__ v16h Afrag[2][4][32];
    __shared__ v16h Bfrag[2][4][32];

    const int tid  = threadIdx.x;
    const int lane = tid & 31;
    const int wave = tid >> 5;
    const int K3   = 3 * Cin;
    const int cmask = Cin - 1;

    v8f acc[4];
    #pragma unroll
    for (int nt = 0; nt < 4; ++nt)
        #pragma unroll
        for (int i = 0; i < 8; ++i) acc[nt][i] = 0.0f;

    const int mBlk = blockIdx.x * 64;
    const int nBlk = blockIdx.y * 64;

    v8h ra[2], rb[2];

    auto loadTile = [&](int k0) {
        #pragma unroll
        for (int gg = 0; gg < 2; ++gg) {
            int g   = tid + gg * 128;
            int r   = g >> 2;
            int cg  = (g & 3) * 8;
            int kk  = k0 + cg;
            int kw  = kk >> lcsh;          // 0..2
            int c0  = kk & cmask;
            int row = mBlk + r;
            int l   = row & (LSEQ - 1);
            int l2  = l + kw - 1;
            if (l2 >= 0 && l2 < LSEQ) {
                int tok = ((row >> LSEQ_LOG) << LSEQ_LOG) + l2;   // b*L + l2
                const float* p = &X[(tok << lcsh) + c0];
                ra[gg] = cvt8(*(const float4*)p, *(const float4*)(p + 4));
            } else {
                ra[gg] = zero8();
            }
        }
        #pragma unroll
        for (int gg = 0; gg < 2; ++gg) {
            int g  = tid + gg * 128;
            int nn = g >> 2;
            int kg = (g & 3) * 8;
            int kk = k0 + kg;
            int kw = kk >> lcsh;
            int c0 = kk & cmask;
            const float* p = &Wc[(nBlk + nn) * K3 + c0 * 3 + kw];
            #pragma unroll
            for (int j = 0; j < 8; ++j) rb[gg][j] = (_Float16)p[3 * j];
        }
    };
    auto storeTile = [&](int buf) {
        #pragma unroll
        for (int gg = 0; gg < 2; ++gg) {
            int g = tid + gg * 128;
            store_afrag_group(Afrag[buf], g >> 2, (g & 3) * 8, ra[gg]);
        }
        #pragma unroll
        for (int gg = 0; gg < 2; ++gg) {
            int g = tid + gg * 128;
            store_bfrag_group(Bfrag[buf], g >> 2, (g & 3) * 8, rb[gg]);
        }
    };

    const int nk = K3 >> 5;
    loadTile(0);
    storeTile(0);
    __syncthreads();

    for (int kt = 0; kt < nk; ++kt) {
        const int cb = kt & 1;
        const bool hasNext = (kt + 1 < nk);
        if (hasNext) loadTile((kt + 1) << 5);

        v16h afrag = Afrag[cb][wave][lane];
        #pragma unroll
        for (int nt = 0; nt < 4; ++nt) {
            v16h bfrag = Bfrag[cb][nt][lane];
            acc[nt] = __builtin_amdgcn_wmma_f32_16x16x32_f16(
                false, afrag, false, bfrag, (short)0, acc[nt], false, false);
        }
        if (hasNext) {
            storeTile(cb ^ 1);
            __syncthreads();
        }
    }

    const int mBase = mBlk + wave * 16 + ((lane >> 4) * 8);
    #pragma unroll
    for (int nt = 0; nt < 4; ++nt) {
        const int n = nBlk + nt * 16 + (lane & 15);
        #pragma unroll
        for (int i = 0; i < 8; ++i) {
            float v = acc[nt][i] * bn_g[n] * escale + bn_b[n];
            C[(mBase + i) * N + n] = v > 0.f ? v : 0.f;
        }
    }
}

// ---------------------------------------------------------------------------
// Depthwise causal conv (D_CONV=4) + SiLU.  xm = first di cols of xz.
// ---------------------------------------------------------------------------
__global__ __launch_bounds__(256)
void mconv_silu_kernel(const float* __restrict__ xz,
                       const float* __restrict__ w, const float* __restrict__ bc,
                       float* __restrict__ xs, int dish)
{
    int idx = blockIdx.x * 256 + threadIdx.x;
    int di  = 1 << dish;
    int ch  = idx & (di - 1);
    int t   = idx >> dish;
    int l   = t & (LSEQ - 1);
    float acc = bc[ch];
    #pragma unroll
    for (int k = 0; k < 4; ++k) {
        int ls = l + k - 3;
        if (ls >= 0)
            acc += w[ch * 4 + k] * xz[((t + (k - 3)) << (dish + 1)) + ch];
    }
    xs[idx] = acc / (1.0f + __expf(-acc));
}

// ---------------------------------------------------------------------------
// Selective scan: h[s] = exp(dt*A[s])*h[s] + dt*u*B[t,s];  y = sum_s h*C[t,s]
// + u*D, then *= silu(z).  One thread per (batch, channel); 16 f32 states in
// registers; B/C broadcast through LDS in 16-step chunks.
// ---------------------------------------------------------------------------
__global__ __launch_bounds__(256)
void scan_kernel(const float* __restrict__ xs, const float* __restrict__ dtp,
                 const float* __restrict__ xdbl, int nxd, int dtr,
                 const float* __restrict__ A_log, const float* __restrict__ Dp,
                 const float* __restrict__ xz, int dish,
                 float* __restrict__ y)
{
    const int b  = blockIdx.x;
    const int ch = blockIdx.y * 256 + threadIdx.x;
    const int di = 1 << dish;

    float As[16], h[16];
    #pragma unroll
    for (int s = 0; s < 16; ++s) {
        As[s] = -__expf(A_log[ch * 16 + s]);
        h[s]  = 0.0f;
    }
    const float Dv = Dp[ch];

    __shared__ float sh[16][32];   // [tt][ 0..15 = B, 16..31 = C ]

    for (int t0 = 0; t0 < LSEQ; t0 += 16) {
        __syncthreads();
        for (int i = threadIdx.x; i < 16 * 32; i += 256) {
            int tt = i >> 5, j = i & 31;
            sh[tt][j] = xdbl[((b << LSEQ_LOG) + t0 + tt) * nxd + dtr + j];
        }
        __syncthreads();
        for (int tt = 0; tt < 16; ++tt) {
            int t  = t0 + tt;
            int ro = (((b << LSEQ_LOG) + t) << dish) + ch;
            float dtv = dtp[ro];
            float u   = xs[ro];
            float du  = dtv * u;
            float yv  = 0.0f;
            #pragma unroll
            for (int s = 0; s < 16; ++s) {
                float dA = __expf(dtv * As[s]);
                h[s] = dA * h[s] + du * sh[tt][s];
                yv  += h[s] * sh[tt][16 + s];
            }
            yv += u * Dv;
            float zv = xz[(((b << LSEQ_LOG) + t) << (dish + 1)) + di + ch];
            yv *= zv / (1.0f + __expf(-zv));
            y[ro] = yv;
        }
    }
}

// ---------------------------------------------------------------------------
// out[t,:] = res[t,:] + LayerNorm(mo[t,:]) * g + b      (one token per block)
// ---------------------------------------------------------------------------
__global__ __launch_bounds__(256)
void ln_add_kernel(const float* __restrict__ mo, const float* __restrict__ res,
                   const float* __restrict__ g, const float* __restrict__ be,
                   float* __restrict__ out, int d)
{
    const int t = blockIdx.x;
    __shared__ float r1[256], r2[256];
    float s = 0.f, s2 = 0.f;
    for (int i = threadIdx.x; i < d; i += 256) {
        float v = mo[t * d + i];
        s += v; s2 += v * v;
    }
    r1[threadIdx.x] = s; r2[threadIdx.x] = s2;
    __syncthreads();
    for (int off = 128; off > 0; off >>= 1) {
        if (threadIdx.x < off) {
            r1[threadIdx.x] += r1[threadIdx.x + off];
            r2[threadIdx.x] += r2[threadIdx.x + off];
        }
        __syncthreads();
    }
    float mean = r1[0] / d;
    float var  = r2[0] / d - mean * mean;
    float inv  = rsqrtf(var + LN_EPS);
    for (int i = threadIdx.x; i < d; i += 256) {
        int idx = t * d + i;
        out[idx] = res[idx] + (mo[idx] - mean) * inv * g[i] + be[i];
    }
}

// ---------------------------------------------------------------------------
extern "C" void kernel_launch(void* const* d_in, const int* in_sizes, int n_in,
                              void* d_out, int out_size, void* d_ws, size_t ws_size,
                              hipStream_t stream)
{
    (void)in_sizes; (void)n_in; (void)out_size; (void)ws_size;

    const int Bb = 4, T = Bb * LSEQ;
    const int outs[2] = {256, 512};
    const float bn_scale = 1.0f / sqrtf(1.0f + BN_EPS);

    const float* x = (const float*)d_in[0];
    float* ws = (float*)d_ws;

    // workspace layout (floats), sized for layer 2 (d=512, di=1024)
    float* resb = ws;                          // T*512
    float* xz   = resb + (size_t)T * 512;      // T*2048
    float* xs   = xz   + (size_t)T * 2048;     // T*1024
    float* xdb  = xs   + (size_t)T * 1024;     // T*64
    float* dtb  = xdb  + (size_t)T * 64;       // T*1024
    float* yb   = dtb  + (size_t)T * 1024;     // T*1024
    float* mo   = yb   + (size_t)T * 1024;     // T*512
    float* cur1 = mo   + (size_t)T * 512;      // T*512

    const float* cur = x;
    int in_sz = 128, in_log = 7;

    for (int layer = 0; layer < 2; ++layer) {
        void* const* P = d_in + 1 + layer * 15;
        const float* conv1_w = (const float*)P[0];
        const float* bn_g    = (const float*)P[1];
        const float* bn_b    = (const float*)P[2];
        const float* skip_w  = (const float*)P[3];
        const float* in_proj = (const float*)P[4];
        const float* mconv_w = (const float*)P[5];
        const float* mconv_b = (const float*)P[6];
        const float* x_proj  = (const float*)P[7];
        const float* dt_w    = (const float*)P[8];
        const float* dt_b    = (const float*)P[9];
        const float* A_log   = (const float*)P[10];
        const float* Dp      = (const float*)P[11];
        const float* out_prj = (const float*)P[12];
        const float* ln_g    = (const float*)P[13];
        const float* ln_b    = (const float*)P[14];

        const int d    = outs[layer];
        const int di   = 2 * d;
        const int dish = (d == 256) ? 9 : 10;   // log2(di)
        const int dtr  = d / 16;
        const int nxd  = dtr + 32;

        dim3 blk(128);

        // 1) resblock conv (im2col WMMA, k-major contraction) + BN + ReLU -> res
        conv_gemm_wmma<<<dim3(T / 64, d / 64), blk, 0, stream>>>(
            cur, in_sz, in_log, conv1_w, bn_g, bn_b, resb, d, bn_scale);

        // 2) skip 1x1 conv, accumulate into res          (aligned K,N)
        gemm_wmma<2, true><<<dim3(T / 64, d / 64), blk, 0, stream>>>(
            cur, in_sz, skip_w, in_sz, resb, d, d, in_sz, nullptr, nullptr, 0.f);

        // 3) in_proj: xz = res @ in_proj^T               (aligned)
        gemm_wmma<0, true><<<dim3(T / 64, (2 * di) / 64), blk, 0, stream>>>(
            resb, d, in_proj, d, xz, 2 * di, 2 * di, d, nullptr, nullptr, 0.f);

        // 4) depthwise causal conv + SiLU -> xs
        mconv_silu_kernel<<<dim3((T * di) / 256), dim3(256), 0, stream>>>(
            xz, mconv_w, mconv_b, xs, dish);

        // 5) x_proj: x_dbl = xs @ x_proj^T               (N=48/64: guarded)
        gemm_wmma<0, false><<<dim3(T / 64, (nxd + 63) / 64), blk, 0, stream>>>(
            xs, di, x_proj, di, xdb, nxd, nxd, di, nullptr, nullptr, 0.f);

        // 6) dt = softplus(x_dbl[:, :dtr] @ dt_w^T + dt_b)  (K=16/32: guarded)
        gemm_wmma<3, false><<<dim3(T / 64, di / 64), blk, 0, stream>>>(
            xdb, nxd, dt_w, dtr, dtb, di, di, dtr, dt_b, nullptr, 0.f);

        // 7) selective scan + D skip + SiLU(z) gate -> y
        scan_kernel<<<dim3(Bb, di / 256), dim3(256), 0, stream>>>(
            xs, dtb, xdb, nxd, dtr, A_log, Dp, xz, dish, yb);

        // 8) out_proj: mo = y @ out_proj^T               (aligned)
        gemm_wmma<0, true><<<dim3(T / 64, d / 64), blk, 0, stream>>>(
            yb, di, out_prj, di, mo, d, d, di, nullptr, nullptr, 0.f);

        // 9) out = res + LN(mo)
        float* dst = (layer == 1) ? (float*)d_out : cur1;
        ln_add_kernel<<<dim3(T), dim3(256), 0, stream>>>(
            mo, resb, ln_g, ln_b, dst, d);

        cur = dst;
        in_sz = d;
        in_log = (d == 256) ? 8 : 9;
    }
}
